// TTS_Simple_67362267071040
// MI455X (gfx1250) — matrix operand order repaired and checked by reading, not compiled, not run on hardware
//
#include <hip/hip_runtime.h>

#define B_    128
#define E_    512
#define H_    1024
#define MEL_  128
#define TTXT  256
#define TMEL  800
#define KE_   (E_ + H_)    // 1536 : encoder [x;h]
#define KD_   (MEL_ + H_)  // 1152 : decoder [mel;h]
#define K3_   (3 * E_)     // 1536 : im2col K

typedef __attribute__((ext_vector_type(16))) __bf16 v16bf;
typedef __attribute__((ext_vector_type(8)))  __bf16 v8bf;
typedef __attribute__((ext_vector_type(8)))  float  v8f;

__device__ inline __bf16 f2bf(float x) {
  union { float f; unsigned u; } v; v.f = x;
  unsigned r = v.u + 0x7fffu + ((v.u >> 16) & 1u);
  unsigned short h = (unsigned short)(r >> 16);
  return __builtin_bit_cast(__bf16, h);
}
__device__ inline float sigm(float x) { return 1.0f / (1.0f + __expf(-x)); }

__device__ inline v16bf join16(v8bf lo, v8bf hi) {
  return __builtin_shufflevector(lo, hi, 0,1,2,3,4,5,6,7,8,9,10,11,12,13,14,15);
}
// A tile 16x32 bf16 (row-major, row stride lda implied by caller's row ptr):
// lane<16 holds row r: K [k0..k0+7] then [k0+16..k0+23]; lane>=16: +8 shift.
__device__ inline v16bf frag_a(const __bf16* arow, int k0, int hi) {
  v8bf lo = *(const v8bf*)(arow + k0 + hi * 8);
  v8bf hh = *(const v8bf*)(arow + k0 + 16 + hi * 8);
  return join16(lo, hh);
}
// B tile 32x16 bf16 from (N,K) row-major weights: lane<16 = col n0+r, K[k0..k0+7] then [k0+16..k0+23];
// lane>=16: +8 shift (same K order as frag_a).
__device__ inline v16bf frag_b(const __bf16* wrow, int k0, int hi) {
  v8bf lo = *(const v8bf*)(wrow + k0 + hi * 8);
  v8bf hh = *(const v8bf*)(wrow + k0 + 16 + hi * 8);
  return join16(lo, hh);
}
__device__ inline v8f wmma_bf(v16bf a, v16bf b, v8f c) {
  return __builtin_amdgcn_wmma_f32_16x16x32_bf16(false, a, false, b, (short)0, c, false, false);
}

// ---------------- weight prep ----------------
__global__ void k_cvt(const float* __restrict__ s, __bf16* __restrict__ d, int n) {
  for (int i = blockIdx.x * blockDim.x + threadIdx.x; i < n; i += gridDim.x * blockDim.x)
    d[i] = f2bf(s[i]);
}
// conv W (O,I,3) -> (O, 3*E) laid as [tap*E + i]
__global__ void k_convw(const float* __restrict__ s, __bf16* __restrict__ d) {
  int o = blockIdx.x;
  for (int idx = threadIdx.x; idx < K3_; idx += blockDim.x) {
    int tap = idx / E_, i = idx % E_;
    d[(size_t)o * K3_ + idx] = f2bf(s[((size_t)o * E_ + i) * 3 + tap]);
  }
}
// concat [Wih (4H,kx) ; Whh (4H,H)] -> (4H, kx+H) bf16
__global__ void k_cat(const float* __restrict__ wih, int kx,
                      const float* __restrict__ whh, __bf16* __restrict__ d) {
  int n = blockIdx.x;
  int Kt = kx + H_;
  for (int k = threadIdx.x; k < Kt; k += blockDim.x) {
    float v = (k < kx) ? wih[(size_t)n * kx + k] : whh[(size_t)n * H_ + (k - kx)];
    d[(size_t)n * Kt + k] = f2bf(v);
  }
}
__global__ void k_init(float* hf, float* cf, __bf16* hb, __bf16* melb) {
  int i = blockIdx.x * blockDim.x + threadIdx.x;
  if (i < B_ * H_) { hf[i] = 0.0f; cf[i] = 0.0f; hb[i] = (__bf16)0.0f; }
  if (i < B_ * MEL_) melb[i] = (__bf16)0.0f;
}
__global__ void k_embed(const int* __restrict__ x, const float* __restrict__ emb,
                        __bf16* __restrict__ out) {
  int idx = blockIdx.x;                 // b*T + t
  const float* row = emb + (size_t)x[idx] * E_;
  __bf16* d = out + (size_t)idx * E_;
  for (int i = threadIdx.x; i < E_; i += blockDim.x) d[i] = f2bf(row[i]);
}
// im2col for k=3, pad=1 along time; row idx = b*T + t
__global__ void k_im2col(const __bf16* __restrict__ src, __bf16* __restrict__ dst) {
  int idx = blockIdx.x;
  int t = idx % TTXT;
  for (int tap = 0; tap < 3; tap++) {
    int ts = t + tap - 1;
    bool ok = (ts >= 0) && (ts < TTXT);
    const __bf16* s = src + ((size_t)idx + (tap - 1)) * E_;
    __bf16* d = dst + (size_t)idx * K3_ + (size_t)tap * E_;
    for (int i = threadIdx.x; i < E_; i += blockDim.x)
      d[i] = ok ? s[i] : (__bf16)0.0f;
  }
}

// ---------------- generic GEMM: C(bf16) = A(M,K)bf16 @ W(N,K)^T + bias [+BN+ReLU] ---------
// block = 256 threads (8 waves), block tile 128x128, wave tile 64x32.
__global__ void k_gemm(const __bf16* __restrict__ A, int lda,
                       const __bf16* __restrict__ W, int ldw,
                       const float* __restrict__ bias,
                       const float* __restrict__ bng, const float* __restrict__ bnb,
                       int bn_relu,
                       __bf16* __restrict__ C, int ldc, int K) {
  const int wave = threadIdx.x >> 5;
  const int lane = threadIdx.x & 31;
  const int r = lane & 15, hi = lane >> 4;
  const int wm = wave >> 2, wn = wave & 3;
  const int m_base = blockIdx.y * 128 + wm * 64;
  const int n_base = blockIdx.x * 128 + wn * 32;

  v8f acc[4][2];
  #pragma unroll
  for (int i = 0; i < 4; i++)
    #pragma unroll
    for (int j = 0; j < 2; j++)
      #pragma unroll
      for (int v = 0; v < 8; v++) acc[i][j][v] = 0.0f;

  for (int k0 = 0; k0 < K; k0 += 32) {
    v16bf bf[2];
    #pragma unroll
    for (int j = 0; j < 2; j++)
      bf[j] = frag_b(W + (size_t)(n_base + j * 16 + r) * ldw, k0, hi);
    #pragma unroll
    for (int i = 0; i < 4; i++) {
      v16bf af = frag_a(A + (size_t)(m_base + i * 16 + r) * lda, k0, hi);
      #pragma unroll
      for (int j = 0; j < 2; j++)
        acc[i][j] = wmma_bf(af, bf[j], acc[i][j]);
    }
  }
  const float rs = rsqrtf(1.0f + 1e-5f);
  #pragma unroll
  for (int i = 0; i < 4; i++) {
    #pragma unroll
    for (int j = 0; j < 2; j++) {
      int col = n_base + j * 16 + r;
      float bcol = bias ? bias[col] : 0.0f;
      #pragma unroll
      for (int v = 0; v < 8; v++) {
        int row = m_base + i * 16 + v + hi * 8;
        float val = acc[i][j][v] + bcol;
        if (bn_relu) { val = bng[col] * (val * rs) + bnb[col]; val = fmaxf(val, 0.0f); }
        C[(size_t)row * ldc + col] = f2bf(val);
      }
    }
  }
}

// ---------------- fused LSTM step: gates GEMM (M=128, N=4H, K=KX+H) + cell update ----------
// grid = H/16 blocks (16 hidden units each), 128 threads = 4 waves, wave w = gate w.
// xbase: row b at xbase + b*xstride (bf16, KX wide). seq_lens!=null -> encoder masking at t.
__global__ void __launch_bounds__(128)
k_lstm_step(const __bf16* __restrict__ xbase, size_t xstride, int KX,
            const __bf16* __restrict__ wcat, int Ktot,
            const float* __restrict__ bias,
            const int* __restrict__ seq_lens, int t,
            const __bf16* __restrict__ h_in_bf, const float* __restrict__ h_in_f,
            const float* __restrict__ c_in,
            __bf16* __restrict__ h_out_bf, float* __restrict__ h_out_f,
            float* __restrict__ c_out) {
  __shared__ float lds[4][B_][16];
  const int wave = threadIdx.x >> 5;
  const int lane = threadIdx.x & 31;
  const int r = lane & 15, hi = lane >> 4;
  const int j0 = blockIdx.x * 16;
  const int n0 = wave * H_ + j0;        // this wave's gate-column base

  v8f acc[8];
  #pragma unroll
  for (int i = 0; i < 8; i++)
    #pragma unroll
    for (int v = 0; v < 8; v++) acc[i][v] = 0.0f;

  const __bf16* wrow = wcat + (size_t)(n0 + r) * Ktot;
  for (int k0 = 0; k0 < Ktot; k0 += 32) {
    v16bf bf = frag_b(wrow, k0, hi);
    const bool in_x = (k0 < KX);        // sections are 32-multiples; tile never straddles
    const int kk = in_x ? k0 : (k0 - KX);
    #pragma unroll
    for (int i = 0; i < 8; i++) {
      int row = i * 16 + r;             // batch index
      const __bf16* ar = in_x ? (xbase + (size_t)row * xstride)
                              : (h_in_bf + (size_t)row * H_);
      v16bf af = frag_a(ar, kk, hi);
      acc[i] = wmma_bf(af, bf, acc[i]);
    }
  }
  float bcol = bias[n0 + r];
  #pragma unroll
  for (int i = 0; i < 8; i++)
    #pragma unroll
    for (int v = 0; v < 8; v++)
      lds[wave][i * 16 + v + hi * 8][r] = acc[i][v] + bcol;
  __syncthreads();

  int b = threadIdx.x;                  // 128 threads = 128 batch rows
  bool valid = seq_lens ? (t < seq_lens[b]) : true;
  #pragma unroll 4
  for (int j = 0; j < 16; j++) {
    int jg = j0 + j;
    float ig = sigm(lds[0][b][j]);
    float fg = sigm(lds[1][b][j]);
    float gg = tanhf(lds[2][b][j]);
    float og = sigm(lds[3][b][j]);
    float cold = c_in[(size_t)b * H_ + jg];
    float c2 = fg * cold + ig * gg;
    float h2 = og * tanhf(c2);
    if (!valid) { c2 = cold; h2 = h_in_f[(size_t)b * H_ + jg]; }
    c_out[(size_t)b * H_ + jg] = c2;
    h_out_f[(size_t)b * H_ + jg] = h2;
    h_out_bf[(size_t)b * H_ + jg] = f2bf(h2);
  }
}

// ---------------- decoder projection: mel = h2 @ Wp^T + bp ; gate = h2 . Wg + bg ----------
// grid = 17 blocks x 128 threads. Blocks 0..15: 64 WMMA tiles (one per wave). Block 16: gate.
__global__ void __launch_bounds__(128)
k_dec_proj(const __bf16* __restrict__ h_bf, const float* __restrict__ h_f,
           const __bf16* __restrict__ wp, const float* __restrict__ bp,
           const float* __restrict__ Wg, const float* __restrict__ bg,
           const int* __restrict__ mel_lens, int s,
           __bf16* __restrict__ mel_bf_out,
           float* __restrict__ out_mel, float* __restrict__ out_gate) {
  if (blockIdx.x == 16) {
    int b = threadIdx.x;
    float acc = bg[0];
    for (int k = 0; k < H_; k++) acc += h_f[(size_t)b * H_ + k] * Wg[k];
    bool m = s > mel_lens[b];
    out_gate[(size_t)b * TMEL + s] = m ? 1000.0f : acc;
    return;
  }
  const int wave = threadIdx.x >> 5;
  const int lane = threadIdx.x & 31;
  const int r = lane & 15, hi = lane >> 4;
  const int tile = blockIdx.x * 4 + wave;     // 0..63
  const int m0 = (tile >> 3) * 16, n0 = (tile & 7) * 16;

  v8f acc;
  #pragma unroll
  for (int v = 0; v < 8; v++) acc[v] = 0.0f;
  const __bf16* wrow = wp + (size_t)(n0 + r) * H_;
  const __bf16* arow = h_bf + (size_t)(m0 + r) * H_;
  for (int k0 = 0; k0 < H_; k0 += 32)
    acc = wmma_bf(frag_a(arow, k0, hi), frag_b(wrow, k0, hi), acc);

  int col = n0 + r;
  float bcol = bp[col];
  #pragma unroll
  for (int v = 0; v < 8; v++) {
    int b = m0 + v + hi * 8;
    float val = acc[v] + bcol;
    mel_bf_out[(size_t)b * MEL_ + col] = f2bf(val);        // unmasked feedback
    bool m = s > mel_lens[b];
    out_mel[((size_t)b * TMEL + s) * MEL_ + col] = m ? 0.0f : val;
  }
}

__global__ void k_mask(const int* __restrict__ lens, float* __restrict__ out_mask) {
  int i = blockIdx.x * blockDim.x + threadIdx.x;
  if (i < B_ * TMEL) {
    int b = i / TMEL, t = i % TMEL;
    out_mask[i] = (t > lens[b]) ? 1.0f : 0.0f;
  }
}

extern "C" void kernel_launch(void* const* d_in, const int* in_sizes, int n_in,
                              void* d_out, int out_size, void* d_ws, size_t ws_size,
                              hipStream_t stream) {
  (void)in_sizes; (void)n_in; (void)out_size; (void)ws_size;
  const int*   x          = (const int*)d_in[0];
  const int*   text_lens  = (const int*)d_in[1];
  /* d_in[2] = y : only shape/dtype used by reference */
  const int*   mel_lens   = (const int*)d_in[3];
  const float* emb        = (const float*)d_in[4];
  const float* enc_lin_W  = (const float*)d_in[5];
  const float* enc_lin_b  = (const float*)d_in[6];
  const float* conv1_W    = (const float*)d_in[7];
  const float* conv1_b    = (const float*)d_in[8];
  const float* bn1_g      = (const float*)d_in[9];
  const float* bn1_b      = (const float*)d_in[10];
  const float* conv2_W    = (const float*)d_in[11];
  const float* conv2_b    = (const float*)d_in[12];
  const float* bn2_g      = (const float*)d_in[13];
  const float* bn2_b      = (const float*)d_in[14];
  const float* Wih_e      = (const float*)d_in[15];
  const float* Whh_e      = (const float*)d_in[16];
  const float* b_e        = (const float*)d_in[17];
  const float* Wih_d      = (const float*)d_in[18];
  const float* Whh_d      = (const float*)d_in[19];
  const float* b_d        = (const float*)d_in[20];
  const float* Wp         = (const float*)d_in[21];
  const float* bp         = (const float*)d_in[22];
  const float* Wg         = (const float*)d_in[23];
  const float* bg         = (const float*)d_in[24];

  float* out_mel  = (float*)d_out;
  float* out_gate = out_mel + (size_t)B_ * TMEL * MEL_;
  float* out_mask = out_gate + (size_t)B_ * TMEL;

  size_t off = 0;
  auto take = [&](size_t bytes) -> void* {
    void* p = (char*)d_ws + off;
    off += (bytes + 255) & ~(size_t)255;
    return p;
  };
  __bf16* w_enclin = (__bf16*)take((size_t)E_ * E_ * 2);
  __bf16* w_conv1  = (__bf16*)take((size_t)E_ * K3_ * 2);
  __bf16* w_conv2  = (__bf16*)take((size_t)E_ * K3_ * 2);
  __bf16* w_cat_e  = (__bf16*)take((size_t)4 * H_ * KE_ * 2);
  __bf16* w_cat_d  = (__bf16*)take((size_t)4 * H_ * KD_ * 2);
  __bf16* w_p      = (__bf16*)take((size_t)MEL_ * H_ * 2);
  __bf16* ebf      = (__bf16*)take((size_t)B_ * TTXT * E_ * 2);  // also conv1 out
  __bf16* h0       = (__bf16*)take((size_t)B_ * TTXT * E_ * 2);  // also conv2 out (xseq)
  __bf16* col      = (__bf16*)take((size_t)B_ * TTXT * K3_ * 2);
  float *hf[2], *cf[2]; __bf16 *hb[2], *mb[2];
  for (int i = 0; i < 2; i++) {
    hf[i] = (float*)take((size_t)B_ * H_ * 4);
    cf[i] = (float*)take((size_t)B_ * H_ * 4);
    hb[i] = (__bf16*)take((size_t)B_ * H_ * 2);
    mb[i] = (__bf16*)take((size_t)B_ * MEL_ * 2);
  }

  // --- weight prep + state init ---
  k_cvt<<<256, 256, 0, stream>>>(enc_lin_W, w_enclin, E_ * E_);
  k_cvt<<<256, 256, 0, stream>>>(Wp, w_p, MEL_ * H_);
  k_convw<<<E_, 256, 0, stream>>>(conv1_W, w_conv1);
  k_convw<<<E_, 256, 0, stream>>>(conv2_W, w_conv2);
  k_cat<<<4 * H_, 256, 0, stream>>>(Wih_e, E_, Whh_e, w_cat_e);
  k_cat<<<4 * H_, 256, 0, stream>>>(Wih_d, MEL_, Whh_d, w_cat_d);
  k_init<<<(B_ * H_ + 255) / 256, 256, 0, stream>>>(hf[0], cf[0], hb[0], mb[0]);

  // --- parallel phase: embed -> enc_lin -> conv1(BN,ReLU) -> conv2(BN,ReLU) ---
  k_embed<<<B_ * TTXT, 128, 0, stream>>>(x, emb, ebf);
  dim3 ggrid(E_ / 128, (B_ * TTXT) / 128);
  k_gemm<<<ggrid, 256, 0, stream>>>(ebf, E_, w_enclin, E_, enc_lin_b,
                                    nullptr, nullptr, 0, h0, E_, E_);
  k_im2col<<<B_ * TTXT, 256, 0, stream>>>(h0, col);
  k_gemm<<<ggrid, 256, 0, stream>>>(col, K3_, w_conv1, K3_, conv1_b,
                                    bn1_g, bn1_b, 1, ebf, E_, K3_);
  k_im2col<<<B_ * TTXT, 256, 0, stream>>>(ebf, col);
  k_gemm<<<ggrid, 256, 0, stream>>>(col, K3_, w_conv2, K3_, conv2_b,
                                    bn2_g, bn2_b, 1, h0, E_, K3_);   // h0 = encoder input seq

  // --- encoder LSTM: 256 sequential fused steps (final state lands in buffer 0) ---
  for (int t = 0; t < TTXT; t++) {
    int a = t & 1, b = (t + 1) & 1;
    k_lstm_step<<<H_ / 16, 128, 0, stream>>>(
        h0 + (size_t)t * E_, (size_t)TTXT * E_, E_, w_cat_e, KE_, b_e,
        text_lens, t, hb[a], hf[a], cf[a], hb[b], hf[b], cf[b]);
  }

  k_mask<<<(B_ * TMEL + 255) / 256, 256, 0, stream>>>(mel_lens, out_mask);

  // --- decoder: 800 sequential (step + projection) pairs ---
  for (int s = 0; s < TMEL; s++) {
    int a = s & 1, b = (s + 1) & 1;
    k_lstm_step<<<H_ / 16, 128, 0, stream>>>(
        mb[a], (size_t)MEL_, MEL_, w_cat_d, KD_, b_d,
        nullptr, 0, hb[a], hf[a], cf[a], hb[b], hf[b], cf[b]);
    k_dec_proj<<<17, 128, 0, stream>>>(hb[b], hf[b], w_p, bp, Wg, bg,
                                       mel_lens, s, mb[b], out_mel, out_gate);
  }
}
